// MultiHeadAttention_56470230007968
// MI455X (gfx1250) — matrix operand order repaired and checked by reading, not compiled, not run on hardware
//
#include <hip/hip_runtime.h>

typedef __attribute__((ext_vector_type(16))) __bf16 v16bf;
typedef __attribute__((ext_vector_type(8)))  float  v8f;

#define D_MODEL   1024
#define NUM_HEADS 16
#define HEAD_DIM  64
#define BATCH     2
#define SEQ       2048
#define M_TOT     (BATCH * SEQ)     /* 4096 rows of x   */
#define N_QKV     (3 * D_MODEL)     /* 3072 qkv outputs */

// ---------------------------------------------------------------------------
// WMMA helper: D(16x16 f32) = A(16x32 bf16) * B(32x16 bf16) + C
// ---------------------------------------------------------------------------
__device__ __forceinline__ v8f wmma_bf16(v16bf a, v16bf b, v8f c) {
  return __builtin_amdgcn_wmma_f32_16x16x32_bf16(
      /*neg_a=*/false, a, /*neg_b=*/false, b,
      /*c_mod=*/(short)0, c, /*reuse_a=*/false, /*reuse_b=*/false);
}

// A fragment: 16x32 tile from row-major memory (leading dim lda).
__device__ __forceinline__ v16bf load_a_frag(const __bf16* base, int lda,
                                             int m0, int k0, int lane) {
  const int l  = lane & 15;
  const int hi = lane >> 4;
  const __bf16* row = base + (size_t)(m0 + l) * lda + k0;
  v16bf a;
#pragma unroll
  for (int e = 0; e < 8; ++e) a[e] = row[hi * 16 + e];
#pragma unroll
  for (int e = 0; e < 8; ++e) a[8 + e] = row[hi * 16 + 8 + e];
  return a;
}

// B fragment (32x16, K x N) where memory holds the transpose:
// element (k, n) = base[(n0+n)*ldb + k0+k]
__device__ __forceinline__ v16bf load_bT_frag(const __bf16* base, int ldb,
                                              int n0, int k0, int lane) {
  const int n    = lane & 15;
  const int koff = (lane >> 4) * 16;
  const __bf16* row = base + (size_t)(n0 + n) * ldb + k0 + koff;
  v16bf b;
#pragma unroll
  for (int e = 0; e < 16; ++e) b[e] = row[e];
  return b;
}

// B fragment (32x16, K x N) from row-major memory:
// element (k, n) = base[(k0+k)*ldb + n0+n]
__device__ __forceinline__ v16bf load_b_frag(const __bf16* base, int ldb,
                                             int k0, int n0, int lane) {
  const int n    = lane & 15;
  const int koff = (lane >> 4) * 16;
  v16bf b;
#pragma unroll
  for (int e = 0; e < 16; ++e)
    b[e] = base[(size_t)(k0 + koff + e) * ldb + n0 + n];
  return b;
}

// Async global->LDS copy of one 16-byte chunk per lane (CDNA5 async path).
// lds_off is the byte offset within the workgroup LDS allocation
// (ISA 10.x); gaddr is a per-lane global addr.
__device__ __forceinline__ void async_copy_b128(unsigned lds_off,
                                                const void* gaddr) {
  asm volatile("global_load_async_to_lds_b128 %0, %1, off"
               :: "v"(lds_off), "v"(gaddr) : "memory");
}

// ---------------------------------------------------------------------------
// fp32 -> bf16 conversion (grid-stride)
// ---------------------------------------------------------------------------
__global__ void cvt_f32_bf16(const float* __restrict__ src,
                             __bf16* __restrict__ dst, int n) {
  int i      = blockIdx.x * blockDim.x + threadIdx.x;
  int stride = gridDim.x * blockDim.x;
  for (; i < n; i += stride) dst[i] = (__bf16)src[i];
}

// ---------------------------------------------------------------------------
// QKV projection: qkv[m][o] = sum_c x[m][c] * w_qkv[o][c] + b_qkv[o]
// scattered into Q/K/V in [B][H][T][Dh] bf16 layout.
// One wave -> one 32x64 output tile (8 WMMAs / k-step, ~21 FLOP/byte);
// k-loop unrolled x2 so the scheduler can pipeline loads across iterations.
// ---------------------------------------------------------------------------
__global__ void qkv_gemm(const __bf16* __restrict__ X,
                         const __bf16* __restrict__ W,
                         const float* __restrict__ bias,
                         __bf16* __restrict__ Qd, __bf16* __restrict__ Kd,
                         __bf16* __restrict__ Vd) {
  const int lane = threadIdx.x & 31;
  const int wave = threadIdx.x >> 5;
  const int TN   = N_QKV / 64;               // 48 tiles along N
  int tile = blockIdx.x * 8 + wave;
  int tm   = tile / TN;
  int tn   = tile - tm * TN;
  if (tm >= M_TOT / 32) return;
  const int m0 = tm * 32, n0 = tn * 64;

  v8f acc[2][4] = {};
#pragma unroll 2
  for (int k0 = 0; k0 < D_MODEL; k0 += 32) {
    if (k0 + 32 < D_MODEL)
      __builtin_prefetch((const void*)(X + (size_t)(m0 + (lane & 15)) * D_MODEL +
                                       k0 + 32), 0, 0);
    v16bf a0 = load_a_frag(X, D_MODEL, m0, k0, lane);
    v16bf a1 = load_a_frag(X, D_MODEL, m0 + 16, k0, lane);
#pragma unroll
    for (int j = 0; j < 4; ++j) {
      v16bf b = load_bT_frag(W, D_MODEL, n0 + j * 16, k0, lane);
      acc[0][j] = wmma_bf16(a0, b, acc[0][j]);
      acc[1][j] = wmma_bf16(a1, b, acc[1][j]);
    }
  }

  const int n  = lane & 15;
  const int hi = lane >> 4;
#pragma unroll
  for (int j = 0; j < 4; ++j) {
    const int o  = n0 + j * 16 + n;
    const float bv = bias[o];
    const int which = o >> 10;        // 0=Q,1=K,2=V
    const int rem   = o & 1023;
    const int h     = rem >> 6;
    const int d     = rem & 63;
    __bf16* dst = (which == 0) ? Qd : (which == 1) ? Kd : Vd;
#pragma unroll
    for (int half = 0; half < 2; ++half) {
#pragma unroll
      for (int r = 0; r < 8; ++r) {
        const int m  = m0 + half * 16 + r + hi * 8;
        const int bb = m >> 11;          // m / SEQ
        const int t  = m & (SEQ - 1);
        dst[((size_t)((bb * NUM_HEADS + h) * SEQ + t)) * HEAD_DIM + d] =
            (__bf16)(acc[half][j][r] + bv);
      }
    }
  }
}

// ---------------------------------------------------------------------------
// Flash attention: one wave per (b, h, 16-row query tile).
// K/V tiles staged into LDS via global_load_async_to_lds_b128 (ASYNCcnt),
// split-wait so Q*K^T overlaps the in-flight V tile.  Online softmax; P goes
// through LDS for the D-layout -> A-layout transpose.
// ---------------------------------------------------------------------------
__global__ void attn_kernel(const __bf16* __restrict__ Q,
                            const __bf16* __restrict__ K,
                            const __bf16* __restrict__ V,
                            __bf16* __restrict__ O) {
  // [0,2048)=K tile 32x64, [2048,4096)=V tile 32x64, [4096,4608)=P 16x32
  __shared__ __bf16 smem[32 * 64 + 32 * 64 + 16 * 32];
  const int lane = threadIdx.x & 31;
  const int qt = blockIdx.x, h = blockIdx.y, b = blockIdx.z;
  const int q0 = qt * 16;
  const size_t bh = (size_t)(b * NUM_HEADS + h) * SEQ * HEAD_DIM;
  const __bf16* Qbh = Q + bh;
  const __bf16* Kbh = K + bh;
  const __bf16* Vbh = V + bh;

  const v16bf qa0 = load_a_frag(Qbh, HEAD_DIM, q0, 0, lane);
  const v16bf qa1 = load_a_frag(Qbh, HEAD_DIM, q0, 32, lane);

  const int n  = lane & 15;
  const int hi = lane >> 4;
  float mrow[8], lrow[8];
  v8f o0 = {}, o1 = {}, o2 = {}, o3 = {};
#pragma unroll
  for (int r = 0; r < 8; ++r) { mrow[r] = -__builtin_inff(); lrow[r] = 0.f; }

  const __bf16* Ks = smem;
  const __bf16* Vs = smem + 2048;
  __bf16*       Ps = smem + 4096;

  for (int kv0 = 0; kv0 <= q0 + 15; kv0 += 32) {
    // ---- async-stage K tile then V tile (each one contiguous 4KB block) ----
    {
      const char* kg = (const char*)(Kbh + (size_t)kv0 * HEAD_DIM);
      const char* vg = (const char*)(Vbh + (size_t)kv0 * HEAD_DIM);
      unsigned off = (unsigned)lane * 16u;
#pragma unroll
      for (int i = 0; i < 8; ++i) { async_copy_b128(off, kg + off); off += 512; }
      off = (unsigned)lane * 16u;
#pragma unroll
      for (int i = 0; i < 8; ++i) {
        async_copy_b128(4096u + off, vg + off);
        off += 512;
      }
      // K tile done when <=8 async ops outstanding (async loads retire in order)
      asm volatile("s_wait_asynccnt 8" ::: "memory");
    }

    // ---- scores S = Q * K^T from LDS ----
    v8f s0 = {}, s1 = {};
    {
      v16bf kb;
      kb = load_bT_frag(Ks, HEAD_DIM, 0, 0, lane);   s0 = wmma_bf16(qa0, kb, s0);
      kb = load_bT_frag(Ks, HEAD_DIM, 0, 32, lane);  s0 = wmma_bf16(qa1, kb, s0);
      kb = load_bT_frag(Ks, HEAD_DIM, 16, 0, lane);  s1 = wmma_bf16(qa0, kb, s1);
      kb = load_bT_frag(Ks, HEAD_DIM, 16, 32, lane); s1 = wmma_bf16(qa1, kb, s1);
    }

    // ---- scale + causal mask + online softmax ----
    const int kg0 = kv0 + n;
    const int kg1 = kv0 + 16 + n;
    float pr0[8], pr1[8];
#pragma unroll
    for (int r = 0; r < 8; ++r) {
      const int qg = q0 + r + hi * 8;
      float v0 = (kg0 <= qg) ? s0[r] * 0.125f : -__builtin_inff();
      float v1 = (kg1 <= qg) ? s1[r] * 0.125f : -__builtin_inff();
      float tmax = fmaxf(v0, v1);
      tmax = fmaxf(tmax, __shfl_xor(tmax, 1, 16));
      tmax = fmaxf(tmax, __shfl_xor(tmax, 2, 16));
      tmax = fmaxf(tmax, __shfl_xor(tmax, 4, 16));
      tmax = fmaxf(tmax, __shfl_xor(tmax, 8, 16));
      const float newm = fmaxf(mrow[r], tmax);
      const float p0 = __expf(v0 - newm);
      const float p1 = __expf(v1 - newm);
      float sum = p0 + p1;
      sum += __shfl_xor(sum, 1, 16);
      sum += __shfl_xor(sum, 2, 16);
      sum += __shfl_xor(sum, 4, 16);
      sum += __shfl_xor(sum, 8, 16);
      const float alpha = __expf(mrow[r] - newm);
      lrow[r] = lrow[r] * alpha + sum;
      mrow[r] = newm;
      o0[r] *= alpha; o1[r] *= alpha; o2[r] *= alpha; o3[r] *= alpha;
      pr0[r] = p0; pr1[r] = p1;
    }

    // ---- D-layout P -> row-major LDS (16x32 bf16) -> A-fragment ----
    __syncthreads();
#pragma unroll
    for (int r = 0; r < 8; ++r) {
      const int row = r + hi * 8;
      Ps[row * 32 + n]      = (__bf16)pr0[r];
      Ps[row * 32 + 16 + n] = (__bf16)pr1[r];
    }
    __syncthreads();
    const v16bf pa = load_a_frag(Ps, 32, 0, 0, lane);

    // ---- wait for V tile, then O += P * V (4 dh tiles of 16) ----
    asm volatile("s_wait_asynccnt 0" ::: "memory");
    v16bf vb;
    vb = load_b_frag(Vs, HEAD_DIM, 0, 0, lane);  o0 = wmma_bf16(pa, vb, o0);
    vb = load_b_frag(Vs, HEAD_DIM, 0, 16, lane); o1 = wmma_bf16(pa, vb, o1);
    vb = load_b_frag(Vs, HEAD_DIM, 0, 32, lane); o2 = wmma_bf16(pa, vb, o2);
    vb = load_b_frag(Vs, HEAD_DIM, 0, 48, lane); o3 = wmma_bf16(pa, vb, o3);
  }

  // ---- epilogue: normalize and write attn output [B*T][C] as bf16 ----
#pragma unroll
  for (int r = 0; r < 8; ++r) {
    const int row = q0 + r + hi * 8;
    const float inv = 1.0f / lrow[r];
    const size_t base =
        ((size_t)(b * SEQ + row)) * D_MODEL + h * HEAD_DIM + n;
    O[base + 0]  = (__bf16)(o0[r] * inv);
    O[base + 16] = (__bf16)(o1[r] * inv);
    O[base + 32] = (__bf16)(o2[r] * inv);
    O[base + 48] = (__bf16)(o3[r] * inv);
  }
}

// ---------------------------------------------------------------------------
// Output projection: out[m][o] = sum_c attn[m][c] * w_fc[o][c] + b_fc[o]
// fp32 output.  One wave -> one 32x64 tile; k-loop unrolled x2.
// ---------------------------------------------------------------------------
__global__ void fc_gemm(const __bf16* __restrict__ A,
                        const __bf16* __restrict__ W,
                        const float* __restrict__ bias,
                        float* __restrict__ out) {
  const int lane = threadIdx.x & 31;
  const int wave = threadIdx.x >> 5;
  const int TN   = D_MODEL / 64;     // 16
  int tile = blockIdx.x * 8 + wave;
  int tm   = tile / TN;
  int tn   = tile - tm * TN;
  if (tm >= M_TOT / 32) return;
  const int m0 = tm * 32, n0 = tn * 64;

  v8f acc[2][4] = {};
#pragma unroll 2
  for (int k0 = 0; k0 < D_MODEL; k0 += 32) {
    if (k0 + 32 < D_MODEL)
      __builtin_prefetch((const void*)(A + (size_t)(m0 + (lane & 15)) * D_MODEL +
                                       k0 + 32), 0, 0);
    v16bf a0 = load_a_frag(A, D_MODEL, m0, k0, lane);
    v16bf a1 = load_a_frag(A, D_MODEL, m0 + 16, k0, lane);
#pragma unroll
    for (int j = 0; j < 4; ++j) {
      v16bf b = load_bT_frag(W, D_MODEL, n0 + j * 16, k0, lane);
      acc[0][j] = wmma_bf16(a0, b, acc[0][j]);
      acc[1][j] = wmma_bf16(a1, b, acc[1][j]);
    }
  }

  const int n  = lane & 15;
  const int hi = lane >> 4;
#pragma unroll
  for (int j = 0; j < 4; ++j) {
    const float bv = bias[n0 + j * 16 + n];
#pragma unroll
    for (int half = 0; half < 2; ++half) {
#pragma unroll
      for (int r = 0; r < 8; ++r) {
        const int m = m0 + half * 16 + r + hi * 8;
        out[(size_t)m * D_MODEL + n0 + j * 16 + n] = acc[half][j][r] + bv;
      }
    }
  }
}

// ---------------------------------------------------------------------------
// Host launch
// ---------------------------------------------------------------------------
extern "C" void kernel_launch(void* const* d_in, const int* in_sizes, int n_in,
                              void* d_out, int out_size, void* d_ws,
                              size_t ws_size, hipStream_t stream) {
  const float* x     = (const float*)d_in[0];
  const float* w_qkv = (const float*)d_in[1];
  const float* b_qkv = (const float*)d_in[2];
  const float* w_fc  = (const float*)d_in[3];
  const float* b_fc  = (const float*)d_in[4];
  float* out = (float*)d_out;

  const size_t NE_X  = (size_t)M_TOT * D_MODEL;     // 4M
  const size_t NE_WQ = (size_t)N_QKV * D_MODEL;     // 3M
  const size_t NE_WF = (size_t)D_MODEL * D_MODEL;   // 1M

  char* ws = (char*)d_ws;
  __bf16* xb    = (__bf16*)ws; ws += NE_X * 2;
  __bf16* wqkvb = (__bf16*)ws; ws += NE_WQ * 2;
  __bf16* wfcb  = (__bf16*)ws; ws += NE_WF * 2;
  __bf16* Qd    = (__bf16*)ws; ws += NE_X * 2;
  __bf16* Kd    = (__bf16*)ws; ws += NE_X * 2;
  __bf16* Vd    = (__bf16*)ws; ws += NE_X * 2;
  __bf16* attnb = (__bf16*)ws; ws += NE_X * 2;      // total 48 MB

  cvt_f32_bf16<<<2048, 256, 0, stream>>>(x, xb, (int)NE_X);
  cvt_f32_bf16<<<2048, 256, 0, stream>>>(w_qkv, wqkvb, (int)NE_WQ);
  cvt_f32_bf16<<<1024, 256, 0, stream>>>(w_fc, wfcb, (int)NE_WF);

  {
    const int tiles = (M_TOT / 32) * (N_QKV / 64);  // 6144
    qkv_gemm<<<tiles / 8, 256, 0, stream>>>(xb, wqkvb, b_qkv, Qd, Kd, Vd);
  }

  attn_kernel<<<dim3(SEQ / 16, NUM_HEADS, BATCH), 32, 0, stream>>>(Qd, Kd, Vd,
                                                                   attnb);

  {
    const int tiles = (M_TOT / 32) * (D_MODEL / 64);  // 2048
    fc_gemm<<<tiles / 8, 256, 0, stream>>>(attnb, wfcb, b_fc, out);
  }
  (void)in_sizes; (void)n_in; (void)out_size; (void)ws_size;
}
